// SatelliteReadingAttention_21457656611312
// MI455X (gfx1250) — compile-verified, matrix operand order checked by reading, not run on hardware
//
#include <hip/hip_runtime.h>
#include <hip/hip_bf16.h>
#include <math.h>

typedef _Float16 half_t;
typedef __attribute__((ext_vector_type(16))) _Float16 v16h;
typedef __attribute__((ext_vector_type(8)))  _Float16 v8h;
typedef __attribute__((ext_vector_type(8)))  float    v8f;

#define NFQ 4096
#define NSK 1024
#define SAT_IN 768
#define FRONT_IN 320
#define MDIM 512
#define NHEADS 8
#define HDIM 64
#define GHIDN 128
#define GMDIM 128
#define GHDIM 16

#if defined(__has_builtin)
#if __has_builtin(__builtin_amdgcn_sched_barrier)
#define SCHED_BARRIER() __builtin_amdgcn_sched_barrier(0)
#endif
#endif
#ifndef SCHED_BARRIER
#define SCHED_BARRIER() do {} while (0)
#endif

// ---------------- WMMA helpers (wave32, 16x16x32 f16 -> f32) ----------------

static __device__ inline v8f frag_zero() {
  v8f z;
#pragma unroll
  for (int i = 0; i < 8; ++i) z[i] = 0.0f;
  return z;
}

static __device__ inline v8f wmma16(v16h a, v16h b, v8f c) {
  return __builtin_amdgcn_wmma_f32_16x16x32_f16(false, a, false, b, (short)0, c,
                                                false, false);
}

// Load a 16x32 fragment from a row-major [16+, ld] f16 array (A-style).
// For B we exploit symmetry: weights stored [N,K] row-major load identically
// (lane indexes N, regs index K) and act as B = W^T (K x N).
static __device__ inline v16h frag_ld(const half_t* __restrict__ base, int ld, int k0) {
  int lane = threadIdx.x & 31;
  const half_t* p = base + (size_t)(lane & 15) * ld + k0 + ((lane >> 4) << 3);
  union { v16h v; v8h h[2]; } u;
  u.h[0] = *(const v8h*)(p);
  u.h[1] = *(const v8h*)(p + 16);
  return u.v;
}

// K=16 real data zero-padded to K=32 (geometric branch, GHD=16).
static __device__ inline v16h frag_ld_pad16(const half_t* __restrict__ base, int ld) {
  int lane = threadIdx.x & 31;
  const half_t* p = base + (size_t)(lane & 15) * ld + ((lane >> 4) << 3);
  union { v16h v; v8h h[2]; } u;
  u.h[0] = *(const v8h*)(p);
#pragma unroll
  for (int i = 0; i < 8; ++i) u.h[1][i] = (_Float16)0.0f;
  return u.v;
}

// Software-pipelined 16-row x (T*16)-col GEMM. A sched_barrier pins the issue
// order (next-step loads BEFORE current-step WMMAs) so the backend cannot
// re-sink loads next to their consumers; waits become partial s_wait_loadcnt
// instead of 0x0, overlapping memory latency with the matrix pipe.
template <int T>
static __device__ void gemm16(const half_t* __restrict__ A,
                              const half_t* __restrict__ Wb, int K, v8f* acc) {
  v16h a_c = frag_ld(A, K, 0);
  v16h b_c[T];
#pragma unroll
  for (int t = 0; t < T; ++t) b_c[t] = frag_ld(Wb + (size_t)t * 16 * K, K, 0);
  for (int k = 0; k < K; k += 32) {
    int kn = k + 32;
    bool more = kn < K;
    v16h a_n = a_c;
    v16h b_n[T];
#pragma unroll
    for (int t = 0; t < T; ++t) b_n[t] = b_c[t];
    if (more) {
      __builtin_prefetch(A + kn + 32, 0, 1);            // global_prefetch_b8
      a_n = frag_ld(A, K, kn);
#pragma unroll
      for (int t = 0; t < T; ++t) b_n[t] = frag_ld(Wb + (size_t)t * 16 * K, K, kn);
    }
    SCHED_BARRIER();                                     // keep loads above WMMAs
#pragma unroll
    for (int t = 0; t < T; ++t) acc[t] = wmma16(a_c, b_c[t], acc[t]);
    SCHED_BARRIER();
    a_c = a_n;
#pragma unroll
    for (int t = 0; t < T; ++t) b_c[t] = b_n[t];
  }
}

// ---------------- LayerNorm stats over a 16 x NC f32 LDS tile ----------------
template <int NC>
static __device__ void ln_stats(const float* tile, float* red, float* red2, float* mrs) {
  const int CPS = NC / 16;
  int tid = threadIdx.x;
  int row = tid & 15, seg = tid >> 4;
  float s = 0.f, ss = 0.f;
#pragma unroll 4
  for (int i = 0; i < CPS; ++i) {
    float v = tile[row * NC + seg * CPS + i];
    s += v; ss += v * v;
  }
  red[row * 16 + seg] = s;
  red2[row * 16 + seg] = ss;
  __syncthreads();
  if (tid < 16) {
    float S = 0.f, SS = 0.f;
    for (int i = 0; i < 16; ++i) { S += red[tid * 16 + i]; SS += red2[tid * 16 + i]; }
    float mean = S / (float)NC;
    float var = SS / (float)NC - mean * mean;
    mrs[tid * 2 + 0] = mean;
    mrs[tid * 2 + 1] = rsqrtf(var + 1e-5f);
  }
  __syncthreads();
}

// ---------------- Small prep kernels ----------------

__global__ __launch_bounds__(256) void convert_kernel(const float* __restrict__ in,
                                                      half_t* __restrict__ out, int n) {
  int i = blockIdx.x * 256 + threadIdx.x;
  if (i < n) out[i] = (half_t)in[i];
}

// front_feat [320, 4096] -> ffT f16 [4096, 320]
__global__ __launch_bounds__(256) void transpose_ff_kernel(const float* __restrict__ ff,
                                                           half_t* __restrict__ out) {
  int i = blockIdx.x * 256 + threadIdx.x;
  if (i >= NFQ * FRONT_IN) return;
  int n = i / FRONT_IN, c = i - n * FRONT_IN;
  out[i] = (half_t)ff[c * NFQ + n];
}

// out[n,m] = silu(sum_j x[n,j]*w[m,j] + b[m]), tiny K (2 or 6)
__global__ __launch_bounds__(256) void mlp1_kernel(const float* __restrict__ x,
                                                   const float* __restrict__ w,
                                                   const float* __restrict__ b,
                                                   half_t* __restrict__ out,
                                                   int M, int N, int K) {
  int i = blockIdx.x * 256 + threadIdx.x;
  if (i >= M * N) return;
  int n = i / N, m = i - n * N;
  float s = b[m];
  for (int j = 0; j < K; ++j) s += x[n * K + j] * w[m * K + j];
  float sig = 1.0f / (1.0f + expf(-s));
  out[i] = (half_t)(s * sig);
}

// ---------------- q_embed = LN( pos + LN(ffT @ faW^T + fa_b) ) ----------------
__global__ __launch_bounds__(256) void qembed_kernel(
    const half_t* __restrict__ ffT, const half_t* __restrict__ h1,
    const half_t* __restrict__ Wfa, const float* __restrict__ fa_b,
    const float* __restrict__ fa_g, const float* __restrict__ fa_bb,
    const half_t* __restrict__ Wpos2, const float* __restrict__ pos_b2,
    const float* __restrict__ qn_g, const float* __restrict__ qn_b,
    half_t* __restrict__ qembed) {
  __shared__ __align__(16) float tile[16 * MDIM];
  __shared__ float red[256], red2[256], mrs[32];
  int tid = threadIdx.x, lane = tid & 31, wv = tid >> 5;
  int row0 = blockIdx.x * 16, col0 = wv * 64;
  int hi = lane >> 4, cn = lane & 15;

  v8f facc[4], pacc[4];
#pragma unroll
  for (int t = 0; t < 4; ++t) { facc[t] = frag_zero(); pacc[t] = frag_zero(); }

  gemm16<4>(ffT + (size_t)row0 * FRONT_IN, Wfa + (size_t)col0 * FRONT_IN, FRONT_IN, facc);
  gemm16<4>(h1 + (size_t)row0 * MDIM, Wpos2 + (size_t)col0 * MDIM, MDIM, pacc);

#pragma unroll
  for (int t = 0; t < 4; ++t)
#pragma unroll
    for (int v = 0; v < 8; ++v) {
      int r = v + 8 * hi, c = col0 + t * 16 + cn;
      tile[r * MDIM + c] = facc[t][v] + fa_b[c];
    }
  __syncthreads();
  ln_stats<MDIM>(tile, red, red2, mrs);
#pragma unroll
  for (int t = 0; t < 4; ++t)
#pragma unroll
    for (int v = 0; v < 8; ++v) {
      int r = v + 8 * hi, c = col0 + t * 16 + cn;
      float fn = (tile[r * MDIM + c] - mrs[r * 2]) * mrs[r * 2 + 1] * fa_g[c] + fa_bb[c];
      tile[r * MDIM + c] = fn + pacc[t][v] + pos_b2[c];
    }
  __syncthreads();
  ln_stats<MDIM>(tile, red, red2, mrs);
  for (int i = tid; i < 16 * MDIM; i += 256) {
    int r = i >> 9, c = i & (MDIM - 1);
    float v = (tile[i] - mrs[r * 2]) * mrs[r * 2 + 1] * qn_g[c] + qn_b[c];
    qembed[(size_t)(row0 + r) * MDIM + c] = (half_t)v;
  }
}

// ---------------- sat_feat = LN(satF @ saW^T + sa_b) ----------------
__global__ __launch_bounds__(256) void gemm_ln_kernel(
    const half_t* __restrict__ A, int K, const half_t* __restrict__ W,
    const float* __restrict__ bias, const float* __restrict__ g,
    const float* __restrict__ bb, half_t* __restrict__ out) {
  __shared__ __align__(16) float tile[16 * MDIM];
  __shared__ float red[256], red2[256], mrs[32];
  int tid = threadIdx.x, lane = tid & 31, wv = tid >> 5;
  int row0 = blockIdx.x * 16, col0 = wv * 64;
  int hi = lane >> 4, cn = lane & 15;

  v8f acc[4];
#pragma unroll
  for (int t = 0; t < 4; ++t) acc[t] = frag_zero();
  gemm16<4>(A + (size_t)row0 * K, W + (size_t)col0 * K, K, acc);

#pragma unroll
  for (int t = 0; t < 4; ++t)
#pragma unroll
    for (int v = 0; v < 8; ++v) {
      int r = v + 8 * hi, c = col0 + t * 16 + cn;
      tile[r * MDIM + c] = acc[t][v] + bias[c];
    }
  __syncthreads();
  ln_stats<MDIM>(tile, red, red2, mrs);
  for (int i = tid; i < 16 * MDIM; i += 256) {
    int r = i >> 9, c = i & (MDIM - 1);
    float v = (tile[i] - mrs[r * 2]) * mrs[r * 2 + 1] * g[c] + bb[c];
    out[(size_t)(row0 + r) * MDIM + c] = (half_t)v;
  }
}

// ---------------- Q/K/V projection (+ fused GeoRoPE, scale, opt. transpose) --
__global__ __launch_bounds__(256) void proj_rope_kernel(
    const half_t* __restrict__ A, const half_t* __restrict__ W,
    const float* __restrict__ bias, const float* __restrict__ xy, float scale,
    int do_rope, int do_transpose, int M, half_t* __restrict__ out) {
  __shared__ __align__(16) float tile[16 * MDIM];
  int tid = threadIdx.x, lane = tid & 31, wv = tid >> 5;
  int row0 = blockIdx.x * 16, col0 = wv * 64;
  int hi = lane >> 4, cn = lane & 15;

  v8f acc[4];
#pragma unroll
  for (int t = 0; t < 4; ++t) acc[t] = frag_zero();
  gemm16<4>(A + (size_t)row0 * MDIM, W + (size_t)col0 * MDIM, MDIM, acc);

#pragma unroll
  for (int t = 0; t < 4; ++t)
#pragma unroll
    for (int v = 0; v < 8; ++v) {
      int r = v + 8 * hi, c = col0 + t * 16 + cn;
      tile[r * MDIM + c] = acc[t][v] + bias[c];
    }
  __syncthreads();

  if (do_rope) {
    // pairs: 16 rows x 8 heads x 16 rotated dims (d in 0..7 with x, 16..23 with y)
    for (int p = tid; p < 16 * 128; p += 256) {
      int r = p >> 7;
      int q = p & 127;
      int h = q >> 4, j = q & 15;
      int d = (j < 8) ? j : (j + 8);                 // 0..7 or 16..23
      int c0 = h * HDIM + d, c1 = c0 + 8;
      float coord = xy[(size_t)(row0 + r) * 2 + (d >= 16 ? 1 : 0)];
      float invf = __powf(10000.0f, -(float)(d & 7) / 8.0f);
      float th = coord * invf;
      float cth = cosf(th), sth = sinf(th);
      float a1 = tile[r * MDIM + c0], a2 = tile[r * MDIM + c1];
      tile[r * MDIM + c0] = a1 * cth - a2 * sth;
      tile[r * MDIM + c1] = a1 * sth + a2 * cth;
    }
    __syncthreads();
  }
  for (int i = tid; i < 16 * MDIM; i += 256) {
    int r = i >> 9, c = i & (MDIM - 1);
    float v = tile[i] * scale;
    if (do_transpose) out[(size_t)c * M + row0 + r] = (half_t)v;
    else out[(size_t)(row0 + r) * MDIM + c] = (half_t)v;
  }
}

// ---------------- geometric branch: LN(hid @ W2^T + b2) @ Wg^T + bg ----------
__global__ __launch_bounds__(256) void geo_kernel(
    const half_t* __restrict__ Hid, const half_t* __restrict__ W2,
    const float* __restrict__ b2, const float* __restrict__ g,
    const float* __restrict__ bb, const half_t* __restrict__ Wg,
    const float* __restrict__ bg, float scale, int M, half_t* __restrict__ out) {
  __shared__ __align__(16) float tile[16 * GHIDN];
  __shared__ __align__(16) half_t hn[16 * GHIDN];
  __shared__ float red[256], red2[256], mrs[32];
  int tid = threadIdx.x, lane = tid & 31, wv = tid >> 5;
  int row0 = blockIdx.x * 16, col0 = wv * 16;   // 8 waves x 16 cols = 128
  int hi = lane >> 4, cn = lane & 15;

  v8f acc[1];
  acc[0] = frag_zero();
  gemm16<1>(Hid + (size_t)row0 * GHIDN, W2 + (size_t)col0 * GHIDN, GHIDN, acc);
#pragma unroll
  for (int v = 0; v < 8; ++v) {
    int r = v + 8 * hi, c = col0 + cn;
    tile[r * GHIDN + c] = acc[0][v] + b2[c];
  }
  __syncthreads();
  ln_stats<GHIDN>(tile, red, red2, mrs);
  for (int i = tid; i < 16 * GHIDN; i += 256) {
    int r = i >> 7, c = i & (GHIDN - 1);
    hn[i] = (half_t)((tile[i] - mrs[r * 2]) * mrs[r * 2 + 1] * g[c] + bb[c]);
  }
  __syncthreads();

  v8f acc2[1];
  acc2[0] = frag_zero();
  gemm16<1>(hn, Wg + (size_t)col0 * GHIDN, GHIDN, acc2);
#pragma unroll
  for (int v = 0; v < 8; ++v) {
    int r = v + 8 * hi, c = col0 + cn;
    out[(size_t)(row0 + r) * GMDIM + c] = (half_t)((acc2[0][v] + bg[c]) * scale);
  }
}

// ---------------- fused attention (flash-style, online softmax) --------------
__global__ __launch_bounds__(256) void attn_kernel(
    const half_t* __restrict__ qr, const half_t* __restrict__ kr,
    const half_t* __restrict__ vT, const half_t* __restrict__ qg,
    const half_t* __restrict__ kg, const float* __restrict__ qxy_g,
    const float* __restrict__ kxy_g, float* __restrict__ out) {
  __shared__ __align__(16) half_t Pb[8][16 * 32];
  __shared__ float Ob[8][16 * 64];
  __shared__ float MS[8][32];
  __shared__ float QXY[32];
  __shared__ float comb[32];
  int tid = threadIdx.x, lane = tid & 31, wv = tid >> 5;
  int row0 = blockIdx.x * 16, h = blockIdx.y;
  int hi = lane >> 4, cn = lane & 15;

  if (tid < 32) QXY[tid] = qxy_g[(size_t)row0 * 2 + tid];
  __syncthreads();

  const half_t* qbase = qr + (size_t)row0 * MDIM + h * HDIM;
  v16h aq0 = frag_ld(qbase, MDIM, 0);
  v16h aq1 = frag_ld(qbase, MDIM, 32);
  v16h ag  = frag_ld_pad16(qg + (size_t)row0 * GMDIM + h * GHDIM, GMDIM);

  float m[8], l[8];
  v8f o[4];
#pragma unroll
  for (int v = 0; v < 8; ++v) { m[v] = -1e30f; l[v] = 0.f; }
#pragma unroll
  for (int t = 0; t < 4; ++t) o[t] = frag_zero();

  float qx[8], qy[8];
#pragma unroll
  for (int v = 0; v < 8; ++v) { int r = v + 8 * hi; qx[v] = QXY[r * 2]; qy[v] = QXY[r * 2 + 1]; }

  for (int chunk = 0; chunk < 4; ++chunk) {
    int kb = wv * 128 + chunk * 32;
    // ---- hoist ALL loads of this chunk (K, geo-K, V, key-xy) before the WMMAs
    const half_t* kb0 = kr + (size_t)kb * MDIM + h * HDIM;
    const half_t* kb1 = kb0 + (size_t)16 * MDIM;
    v16h f_k0a = frag_ld(kb0, MDIM, 0);
    v16h f_k0b = frag_ld(kb0, MDIM, 32);
    v16h f_k1a = frag_ld(kb1, MDIM, 0);
    v16h f_k1b = frag_ld(kb1, MDIM, 32);
    v16h f_g0 = frag_ld_pad16(kg + (size_t)kb * GMDIM + h * GHDIM, GMDIM);
    v16h f_g1 = frag_ld_pad16(kg + (size_t)(kb + 16) * GMDIM + h * GHDIM, GMDIM);
    v16h f_v[4];
#pragma unroll
    for (int t = 0; t < 4; ++t)
      f_v[t] = frag_ld(vT + (size_t)(h * HDIM + t * 16) * NSK, NSK, kb);
    float kx0 = kxy_g[(kb + cn) * 2], ky0 = kxy_g[(kb + cn) * 2 + 1];
    float kx1 = kxy_g[(kb + 16 + cn) * 2], ky1 = kxy_g[(kb + 16 + cn) * 2 + 1];
    SCHED_BARRIER();                                  // loads stay above WMMAs

    v8f c0 = frag_zero(), c1 = frag_zero();
    c0 = wmma16(aq0, f_k0a, c0);
    c0 = wmma16(aq1, f_k0b, c0);
    c0 = wmma16(ag, f_g0, c0);
    c1 = wmma16(aq0, f_k1a, c1);
    c1 = wmma16(aq1, f_k1b, c1);
    c1 = wmma16(ag, f_g1, c1);

    // subtract LAMBDA_GEO * dist2
#pragma unroll
    for (int v = 0; v < 8; ++v) {
      float dx = qx[v] - kx0, dy = qy[v] - ky0;
      c0[v] -= dx * dx + dy * dy;
      dx = qx[v] - kx1; dy = qy[v] - ky1;
      c1[v] -= dx * dx + dy * dy;
    }
    // online softmax over this 32-key chunk
    float mx[8];
#pragma unroll
    for (int v = 0; v < 8; ++v) mx[v] = fmaxf(c0[v], c1[v]);
#pragma unroll
    for (int d = 1; d < 16; d <<= 1)
#pragma unroll
      for (int v = 0; v < 8; ++v) mx[v] = fmaxf(mx[v], __shfl_xor(mx[v], d));
    float f[8];
#pragma unroll
    for (int v = 0; v < 8; ++v) {
      float mn = fmaxf(m[v], mx[v]);
      f[v] = __expf(m[v] - mn);
      m[v] = mn;
      c0[v] = __expf(c0[v] - mn);
      c1[v] = __expf(c1[v] - mn);
    }
    float s[8];
#pragma unroll
    for (int v = 0; v < 8; ++v) s[v] = c0[v] + c1[v];
#pragma unroll
    for (int d = 1; d < 16; d <<= 1)
#pragma unroll
      for (int v = 0; v < 8; ++v) s[v] += __shfl_xor(s[v], d);
#pragma unroll
    for (int v = 0; v < 8; ++v) l[v] = l[v] * f[v] + s[v];
#pragma unroll
    for (int t = 0; t < 4; ++t)
#pragma unroll
      for (int v = 0; v < 8; ++v) o[t][v] *= f[v];
    // P -> LDS (f16), then P @ V (V fragments already in registers)
#pragma unroll
    for (int v = 0; v < 8; ++v) {
      int r = v + 8 * hi;
      Pb[wv][r * 32 + cn] = (half_t)c0[v];
      Pb[wv][r * 32 + 16 + cn] = (half_t)c1[v];
    }
    v16h pa = frag_ld(Pb[wv], 32, 0);
#pragma unroll
    for (int t = 0; t < 4; ++t)
      o[t] = wmma16(pa, f_v[t], o[t]);
  }

  // spill per-wave partials
#pragma unroll
  for (int v = 0; v < 8; ++v) {
    int r = v + 8 * hi;
    if (cn == 0) { MS[wv][r * 2] = m[v]; MS[wv][r * 2 + 1] = l[v]; }
  }
#pragma unroll
  for (int t = 0; t < 4; ++t)
#pragma unroll
    for (int v = 0; v < 8; ++v) {
      int r = v + 8 * hi;
      Ob[wv][r * 64 + t * 16 + cn] = o[t][v];
    }
  __syncthreads();
  if (tid < 16) {
    float M_ = -1e30f;
    for (int w8 = 0; w8 < 8; ++w8) M_ = fmaxf(M_, MS[w8][tid * 2]);
    float S = 0.f;
    for (int w8 = 0; w8 < 8; ++w8) S += __expf(MS[w8][tid * 2] - M_) * MS[w8][tid * 2 + 1];
    comb[tid * 2] = M_;
    comb[tid * 2 + 1] = S;
  }
  __syncthreads();
  for (int i = tid; i < 1024; i += 256) {
    int r = i >> 6, c = i & 63;
    float M_ = comb[r * 2], S = comb[r * 2 + 1];
    float acc = 0.f;
    for (int w8 = 0; w8 < 8; ++w8) acc += __expf(MS[w8][r * 2] - M_) * Ob[w8][r * 64 + c];
    out[(size_t)(row0 + r) * MDIM + h * HDIM + c] = acc / S;
  }
}

// ---------------- host launcher ----------------
extern "C" void kernel_launch(void* const* d_in, const int* in_sizes, int n_in,
                              void* d_out, int out_size, void* d_ws, size_t ws_size,
                              hipStream_t stream) {
  const float* front_feat = (const float*)d_in[0];
  const float* front_xy   = (const float*)d_in[1];
  const float* sat_xy     = (const float*)d_in[3];
  const float* plucker    = (const float*)d_in[4];

  half_t* W = (half_t*)d_ws;
  size_t off = 0;
  auto take = [&](size_t n) { half_t* p = W + off; off += (n + 7) & ~(size_t)7; return p; };
  half_t* ffT      = take((size_t)NFQ * FRONT_IN);
  half_t* satF     = take((size_t)NSK * SAT_IN);
  half_t* h1       = take((size_t)NFQ * MDIM);
  half_t* plk_h    = take((size_t)NFQ * GHIDN);
  half_t* sxy_h    = take((size_t)NSK * GHIDN);
  half_t* q_embed  = take((size_t)NFQ * MDIM);
  half_t* sat_feat = take((size_t)NSK * MDIM);
  half_t* qrb      = take((size_t)NFQ * MDIM);
  half_t* krb      = take((size_t)NSK * MDIM);
  half_t* vT       = take((size_t)MDIM * NSK);
  half_t* qgb      = take((size_t)NFQ * GMDIM);
  half_t* kgb      = take((size_t)NSK * GMDIM);
  half_t* w_pos2   = take((size_t)MDIM * MDIM);
  half_t* w_fa     = take((size_t)MDIM * FRONT_IN);
  half_t* w_sa     = take((size_t)MDIM * SAT_IN);
  half_t* w_q      = take((size_t)MDIM * MDIM);
  half_t* w_k      = take((size_t)MDIM * MDIM);
  half_t* w_v      = take((size_t)MDIM * MDIM);
  half_t* w_plk2   = take((size_t)GHIDN * GHIDN);
  half_t* w_sxy2   = take((size_t)GHIDN * GHIDN);
  half_t* w_qg     = take((size_t)GMDIM * GHIDN);
  half_t* w_kg     = take((size_t)GMDIM * GHIDN);
  (void)ws_size; (void)n_in; (void)in_sizes; (void)out_size;

  auto cvt = [&](const void* src, half_t* dst, int n) {
    convert_kernel<<<dim3((n + 255) / 256), dim3(256), 0, stream>>>((const float*)src, dst, n);
  };
  cvt(d_in[7],  w_pos2, MDIM * MDIM);
  cvt(d_in[9],  w_fa,   MDIM * FRONT_IN);
  cvt(d_in[15], w_sa,   MDIM * SAT_IN);
  cvt(d_in[19], w_q,    MDIM * MDIM);
  cvt(d_in[21], w_k,    MDIM * MDIM);
  cvt(d_in[23], w_v,    MDIM * MDIM);
  cvt(d_in[27], w_plk2, GHIDN * GHIDN);
  cvt(d_in[31], w_sxy2, GHIDN * GHIDN);
  cvt(d_in[37], w_qg,   GMDIM * GHIDN);
  cvt(d_in[39], w_kg,   GMDIM * GHIDN);
  cvt(d_in[2],  satF,   NSK * SAT_IN);

  transpose_ff_kernel<<<dim3((NFQ * FRONT_IN + 255) / 256), dim3(256), 0, stream>>>(front_feat, ffT);

  mlp1_kernel<<<dim3((NFQ * MDIM + 255) / 256), dim3(256), 0, stream>>>(
      front_xy, (const float*)d_in[5], (const float*)d_in[6], h1, NFQ, MDIM, 2);
  mlp1_kernel<<<dim3((NFQ * GHIDN + 255) / 256), dim3(256), 0, stream>>>(
      plucker, (const float*)d_in[25], (const float*)d_in[26], plk_h, NFQ, GHIDN, 6);
  mlp1_kernel<<<dim3((NSK * GHIDN + 255) / 256), dim3(256), 0, stream>>>(
      sat_xy, (const float*)d_in[29], (const float*)d_in[30], sxy_h, NSK, GHIDN, 2);

  qembed_kernel<<<dim3(NFQ / 16), dim3(256), 0, stream>>>(
      ffT, h1, w_fa, (const float*)d_in[10], (const float*)d_in[11], (const float*)d_in[12],
      w_pos2, (const float*)d_in[8], (const float*)d_in[13], (const float*)d_in[14], q_embed);

  gemm_ln_kernel<<<dim3(NSK / 16), dim3(256), 0, stream>>>(
      satF, SAT_IN, w_sa, (const float*)d_in[16], (const float*)d_in[17],
      (const float*)d_in[18], sat_feat);

  const float inv_sqrt_hd = 0.125f;    // 1/sqrt(64) folded into qr
  const float geo_scale   = 0.25f;     // 1/sqrt(16) folded into qg
  proj_rope_kernel<<<dim3(NFQ / 16), dim3(256), 0, stream>>>(
      q_embed, w_q, (const float*)d_in[20], front_xy, inv_sqrt_hd, 1, 0, NFQ, qrb);
  proj_rope_kernel<<<dim3(NSK / 16), dim3(256), 0, stream>>>(
      sat_feat, w_k, (const float*)d_in[22], sat_xy, 1.0f, 1, 0, NSK, krb);
  proj_rope_kernel<<<dim3(NSK / 16), dim3(256), 0, stream>>>(
      sat_feat, w_v, (const float*)d_in[24], nullptr, 1.0f, 0, 1, NSK, vT);

  geo_kernel<<<dim3(NFQ / 16), dim3(256), 0, stream>>>(
      plk_h, w_plk2, (const float*)d_in[28], (const float*)d_in[33], (const float*)d_in[34],
      w_qg, (const float*)d_in[38], geo_scale, NFQ, qgb);
  geo_kernel<<<dim3(NSK / 16), dim3(256), 0, stream>>>(
      sxy_h, w_sxy2, (const float*)d_in[32], (const float*)d_in[35], (const float*)d_in[36],
      w_kg, (const float*)d_in[40], 1.0f, NSK, kgb);

  attn_kernel<<<dim3(NFQ / 16, NHEADS), dim3(256), 0, stream>>>(
      qrb, krb, vT, qgb, kgb, front_xy, sat_xy, (float*)d_out);
}
